// UberCRSN_14929306321452
// MI455X (gfx1250) — matrix operand order repaired and checked by reading, not compiled, not run on hardware
//
#include <hip/hip_runtime.h>
#include <hip/hip_bf16.h>
#include <math.h>

#define EPS 1e-6f
#define CC_W 0.25f
#define ETH_W 0.005f
#define PONDER_W 0.0001f

static constexpr int Bsz  = 1024;   // batch
static constexpr int DIM  = 512;
static constexpr int D2   = 1024;   // 2*DIM
static constexpr int Vv   = 32000;
static constexpr int KCB  = 1024;   // codebook entries
static constexpr int STK  = 16;
static constexpr int TT   = 8;

typedef __attribute__((ext_vector_type(16))) _Float16 v16h;
typedef __attribute__((ext_vector_type(8)))  _Float16 v8h;
typedef __attribute__((ext_vector_type(8)))  float    v8f;

// ---------------------------------------------------------------------------
// CDNA5 async memory->LDS helpers (ASYNCcnt-tracked, bypasses VGPRs)
// ---------------------------------------------------------------------------
typedef __attribute__((address_space(3))) const _Float16 lds_cf16_t;

static __device__ inline unsigned lds_addr_of(const _Float16* p) {
    // generic -> LDS addrspacecast; AS(3) pointers are byte offsets into LDS
    return (unsigned)(unsigned long long)(lds_cf16_t*)p;
}

static __device__ inline void async_load_lds_b128(unsigned lds_byte_addr,
                                                  const _Float16* gaddr) {
    asm volatile("global_load_async_to_lds_b128 %0, %1, off"
                 :: "v"(lds_byte_addr), "v"(gaddr)
                 : "memory");
}

static __device__ inline void wait_async0() {
    asm volatile("s_wait_asynccnt 0" ::: "memory");
}

// ---------------------------------------------------------------------------
// WMMA GEMM:  C[M,N] (f32) = alpha * A[M,K](f32) x B[N,K](f16)^T  (+C) (+bias)
// Tile 128x128x32, 256 threads = 8 wave32 (2x4), 8 WMMA/wave/k-step.
// A staged sync (f32->f16 convert in VGPRs); B staged via async-to-LDS.
// ---------------------------------------------------------------------------
#define BM 128
#define BN 128
#define BK 32
#define LSH 48   // LDS row stride in halves (96B, 32B-aligned fragments)

__global__ __launch_bounds__(256)
void k_gemm(float* __restrict__ C, int ldc,
            const float* __restrict__ A, int lda,
            const _Float16* __restrict__ Bh, int ldb,
            int M, int N, int K,
            float alpha, int accum, const float* __restrict__ bias)
{
    __shared__ _Float16 sA[2][BM * LSH];
    __shared__ _Float16 sB[2][BN * LSH];

    const int tid  = threadIdx.x;
    const int wave = tid >> 5;
    const int lane = tid & 31;
    const int l16  = lane & 15;
    const int hl   = lane >> 4;          // half-wave select

    const int m0 = blockIdx.y * BM;
    const int n0 = blockIdx.x * BN;
    const int wm = (wave >> 2) * 64;     // 0 / 64
    const int wn = (wave & 3) * 32;      // 0..96

    const int srow = tid >> 2;           // 0..63 (staging row)
    const int scol = (tid & 3) * 8;      // 0..24 (staging col, halves)

    v8f acc[4][2] = {};
    const int nk = K / BK;

    auto stage = [&](int buf, int k0) {
#pragma unroll
        for (int i = 0; i < 2; ++i) {
            const int r = srow + i * 64;
            // A: f32 -> f16 convert into LDS (needs VGPR path)
            const float4* ap = (const float4*)(A + (size_t)(m0 + r) * lda + k0 + scol);
            float4 a0 = ap[0], a1 = ap[1];
            _Float16* da = &sA[buf][r * LSH + scol];
            da[0]=(_Float16)a0.x; da[1]=(_Float16)a0.y; da[2]=(_Float16)a0.z; da[3]=(_Float16)a0.w;
            da[4]=(_Float16)a1.x; da[5]=(_Float16)a1.y; da[6]=(_Float16)a1.z; da[7]=(_Float16)a1.w;
            // B: f16, 16B per lane, async DMA memory -> LDS (no VGPR data)
            async_load_lds_b128(lds_addr_of(&sB[buf][r * LSH + scol]),
                                Bh + (size_t)(n0 + r) * ldb + k0 + scol);
        }
    };

    stage(0, 0);
    wait_async0();
    __syncthreads();

    for (int kt = 0; kt < nk; ++kt) {
        const int cur = kt & 1;
        if (kt + 1 < nk) {
            if (kt + 2 < nk)
                __builtin_prefetch(A + (size_t)(m0 + srow) * lda + (kt + 2) * BK, 0, 0);
            stage(cur ^ 1, (kt + 1) * BK);
        }

        v16h af[4], bf[2];
#pragma unroll
        for (int fm = 0; fm < 4; ++fm) {
            const _Float16* ap = &sA[cur][(wm + fm * 16 + l16) * LSH + hl * 16];
            v8h lo = *(const v8h*)ap;          // K = base .. base+7
            v8h hi = *(const v8h*)(ap + 8);    // K = base+8 .. base+15
#pragma unroll
            for (int e = 0; e < 8; ++e) { af[fm][e] = lo[e]; af[fm][e + 8] = hi[e]; }
        }
#pragma unroll
        for (int fn = 0; fn < 2; ++fn) {
            const _Float16* bp = &sB[cur][(wn + fn * 16 + l16) * LSH + hl * 16];
            bf[fn] = *(const v16h*)bp;         // 16 contiguous K halves
        }

#pragma unroll
        for (int fm = 0; fm < 4; ++fm)
#pragma unroll
            for (int fn = 0; fn < 2; ++fn)
                acc[fm][fn] = __builtin_amdgcn_wmma_f32_16x16x32_f16(
                    false, af[fm], false, bf[fn], (short)0, acc[fm][fn], false, false);

        wait_async0();      // async B-tile for `next` visible before barrier
        __syncthreads();
    }

#pragma unroll
    for (int fm = 0; fm < 4; ++fm) {
#pragma unroll
        for (int fn = 0; fn < 2; ++fn) {
            const int col  = n0 + wn + fn * 16 + l16;
            const int row0 = m0 + wm + fm * 16 + hl * 8;
#pragma unroll
            for (int r = 0; r < 8; ++r) {
                float v = acc[fm][fn][r] * alpha;
                if (bias) v += bias[col];
                float* cp = C + (size_t)(row0 + r) * ldc + col;
                if (accum) *cp += v; else *cp = v;
            }
        }
    }
}

// ---------------------------------------------------------------------------
// helpers
// ---------------------------------------------------------------------------
__device__ inline float blockReduceSum(float v, float* sh) {
    const int t = threadIdx.x;
    sh[t] = v; __syncthreads();
    for (int s = 128; s > 0; s >>= 1) { if (t < s) sh[t] += sh[t + s]; __syncthreads(); }
    float r = sh[0]; __syncthreads();
    return r;
}

__global__ void k_convert_h(const float* __restrict__ src, _Float16* __restrict__ dst, long n) {
    long t = (long)blockIdx.x * 256 + threadIdx.x;
    if (t < n) dst[t] = (_Float16)src[t];
}

// src [R][C] f32 -> dst [C][R] f16 ; R,C multiples of 32
__global__ void k_transpose_h(const float* __restrict__ src, _Float16* __restrict__ dst,
                              int R, int C) {
    __shared__ _Float16 tile[32][33];
    const int c0 = blockIdx.x * 32, r0 = blockIdx.y * 32;
    for (int i = threadIdx.y; i < 32; i += 8)
        tile[i][threadIdx.x] = (_Float16)src[(size_t)(r0 + i) * C + c0 + threadIdx.x];
    __syncthreads();
    for (int i = threadIdx.y; i < 32; i += 8)
        dst[(size_t)(c0 + i) * R + r0 + threadIdx.x] = tile[threadIdx.x][i];
}

__global__ void k_embed(const int* __restrict__ ids, const float* __restrict__ mag,
                        const float* __restrict__ ph, float* __restrict__ zr,
                        float* __restrict__ zi) {
    int t = blockIdx.x * 256 + threadIdx.x;      // B*DIM
    int b = t >> 9, d = t & 511;
    int idx = ids[b];
    float r = mag[(size_t)idx * DIM + d], th = ph[(size_t)idx * DIM + d];
    zr[t] = r * __cosf(th);
    zi[t] = r * __sinf(th);
}

__global__ void k_init_state(float* __restrict__ ptr, float* __restrict__ remain) {
    int b = blockIdx.x * 256 + threadIdx.x;
    if (b < Bsz) { ptr[b * STK] = 1.0f; remain[b] = 1.0f; }
}

__global__ void k_cnorm_modrelu(float* __restrict__ pr, float* __restrict__ pi,
                                const float* __restrict__ scl, const float* __restrict__ shf,
                                const float* __restrict__ mbias) {
    __shared__ float sh[256];
    const int b = blockIdx.x;
    float* zr = pr + (size_t)b * DIM;
    float* zi = pi + (size_t)b * DIM;
    float s1 = 0.f, s2 = 0.f;
    for (int d = threadIdx.x; d < DIM; d += 256) {
        float a = zr[d], c = zi[d];
        float m = sqrtf(a * a + c * c) + EPS;
        s1 += m; s2 += m * m;
    }
    s1 = blockReduceSum(s1, sh);
    s2 = blockReduceSum(s2, sh);
    const float mean = s1 / (float)DIM;
    const float var  = (s2 - (float)DIM * mean * mean) / (float)(DIM - 1);
    const float inv  = rsqrtf(var + EPS);
    for (int d = threadIdx.x; d < DIM; d += 256) {
        float a = zr[d], c = zi[d];
        float rr = sqrtf(a * a + c * c);
        float nm = (rr + EPS - mean) * inv * scl[d] + shf[d];
        float ir = rr > 1e-30f ? 1.0f / rr : 0.0f;
        float xr = nm * a * ir, xi = nm * c * ir;
        float n2 = sqrtf(xr * xr + xi * xi) + EPS;
        float tt = n2 + mbias[d];
        float s  = (tt > 0.0f ? tt : 0.0f) / n2;
        zr[d] = xr * s; zi[d] = xi * s;
    }
}

__global__ void k_softmax(float* __restrict__ S, int n, float scl) {
    __shared__ float sh[256];
    float* row = S + (size_t)blockIdx.x * n;
    float mx = -3.4e38f;
    for (int j = threadIdx.x; j < n; j += 256) mx = fmaxf(mx, row[j] * scl);
    sh[threadIdx.x] = mx; __syncthreads();
    for (int s = 128; s > 0; s >>= 1) {
        if (threadIdx.x < s) sh[threadIdx.x] = fmaxf(sh[threadIdx.x], sh[threadIdx.x + s]);
        __syncthreads();
    }
    mx = sh[0]; __syncthreads();
    float sum = 0.f;
    for (int j = threadIdx.x; j < n; j += 256) {
        float e = __expf(row[j] * scl - mx);
        row[j] = e; sum += e;
    }
    sum = blockReduceSum(sum, sh);
    const float iv = 1.0f / sum;
    for (int j = threadIdx.x; j < n; j += 256) row[j] *= iv;
}

__global__ void k_halt_ctrl(const float* __restrict__ zf,
                            const float* __restrict__ hW, const float* __restrict__ hb,
                            const float* __restrict__ cW, const float* __restrict__ cbv,
                            float* __restrict__ halt, float* __restrict__ ctrl) {
    __shared__ float sh[256];
    const int b = blockIdx.x;
    const float* z = zf + (size_t)b * D2;
    float d0 = 0, d1 = 0, d2 = 0, d3 = 0;
    for (int k = threadIdx.x; k < D2; k += 256) {
        float zv = z[k];
        d0 += zv * hW[k];
        d1 += zv * cW[k];
        d2 += zv * cW[D2 + k];
        d3 += zv * cW[2 * D2 + k];
    }
    d0 = blockReduceSum(d0, sh); d1 = blockReduceSum(d1, sh);
    d2 = blockReduceSum(d2, sh); d3 = blockReduceSum(d3, sh);
    if (threadIdx.x == 0) {
        halt[b] = 1.0f / (1.0f + __expf(-(d0 + hb[0])));
        float c0 = d1 + cbv[0], c1 = d2 + cbv[1], c2 = d3 + cbv[2];
        float m  = fmaxf(c0, fmaxf(c1, c2));
        float e0 = __expf(c0 - m), e1 = __expf(c1 - m), e2 = __expf(c2 - m);
        float iv = 1.0f / (e0 + e1 + e2);
        ctrl[b * 3 + 0] = e0 * iv; ctrl[b * 3 + 1] = e1 * iv; ctrl[b * 3 + 2] = e2 * iv;
    }
}

__global__ void k_wrem(const float* __restrict__ halt, float* __restrict__ remain,
                       float* __restrict__ wvec, float* __restrict__ ponder_sum, int last) {
    int b = blockIdx.x * 256 + threadIdx.x;
    if (b < Bsz) {
        float h = halt[b], rm = remain[b];
        wvec[b] = last ? rm : h * rm;
        rm *= (1.0f - h);
        remain[b] = rm;
        atomicAdd(ponder_sum, rm * (1.0f / (float)Bsz));
    }
}

__global__ void k_stack_ptr(float* __restrict__ ptr, const float* __restrict__ ctrl,
                            float* __restrict__ nptr, float* __restrict__ wm) {
    int b = blockIdx.x * 256 + threadIdx.x;
    if (b >= Bsz) return;
    float p[STK], up[STK];
    for (int s = 0; s < STK; ++s) p[s] = ptr[b * STK + s];
    const float push = ctrl[b * 3], pop = ctrl[b * 3 + 1], noop = ctrl[b * 3 + 2];
    float sum = 0.f, np[STK];
    for (int s = 0; s < STK; ++s) {
        up[s] = p[(s + STK - 1) & (STK - 1)];
        float dn = p[(s + 1) & (STK - 1)];
        np[s] = push * up[s] + pop * dn + noop * p[s];
        sum += np[s];
    }
    const float iv = 1.0f / (sum + EPS);
    for (int s = 0; s < STK; ++s) {
        float v = np[s] * iv;
        nptr[b * STK + s] = v;
        wm[b * STK + s]   = push * up[s];
        ptr[b * STK + s]  = v;
    }
}

__global__ void k_stack_mem(float* __restrict__ mem, const float* __restrict__ wm,
                            const float* __restrict__ nptr, const float* __restrict__ zf,
                            float* __restrict__ rd) {
    int t = blockIdx.x * 256 + threadIdx.x;   // B*D2
    int b = t >> 10, d = t & (D2 - 1);
    const float z = zf[(size_t)b * D2 + d];
    float acc = 0.f;
    for (int s = 0; s < STK; ++s) {
        size_t mi = ((size_t)b * STK + s) * D2 + d;
        float w = wm[b * STK + s];
        float m = mem[mi];
        m = w * z + m * (1.0f - w);
        mem[mi] = m;
        acc += m * nptr[b * STK + s];
    }
    rd[(size_t)b * D2 + d] = acc;
}

__global__ void k_cf(const float* __restrict__ zf, const float* __restrict__ rd,
                     float* __restrict__ cf) {
    size_t t = (size_t)blockIdx.x * 256 + threadIdx.x;
    cf[t] = zf[t] + rd[t];
}

__global__ void k_cbnorm(const float* __restrict__ cb, float* __restrict__ cbn) {
    __shared__ float sh[256];
    const int j = blockIdx.x;
    float s = 0.f;
    for (int k = threadIdx.x; k < D2; k += 256) {
        float v = cb[(size_t)j * D2 + k];
        s += v * v;
    }
    s = blockReduceSum(s, sh);
    if (threadIdx.x == 0) cbn[j] = s;
}

__global__ void k_argmin(const float* __restrict__ G, const float* __restrict__ cbn,
                         int* __restrict__ sym) {
    __shared__ float sv[256]; __shared__ int si[256];
    const int b = blockIdx.x;
    float best = 3.4e38f; int bi = 0x7fffffff;
    for (int j = threadIdx.x; j < KCB; j += 256) {
        float s = cbn[j] - 2.0f * G[(size_t)b * KCB + j];
        if (s < best) { best = s; bi = j; }
    }
    sv[threadIdx.x] = best; si[threadIdx.x] = bi; __syncthreads();
    for (int s = 128; s > 0; s >>= 1) {
        if (threadIdx.x < s) {
            float ov = sv[threadIdx.x + s]; int oi = si[threadIdx.x + s];
            if (ov < sv[threadIdx.x] || (ov == sv[threadIdx.x] && oi < si[threadIdx.x])) {
                sv[threadIdx.x] = ov; si[threadIdx.x] = oi;
            }
        }
        __syncthreads();
    }
    if (threadIdx.x == 0) sym[b] = si[0];
}

__global__ void k_eth(const float* __restrict__ adj, const int* __restrict__ sym,
                      int* __restrict__ prev, float* __restrict__ eth_sum, int step) {
    __shared__ float sh[256];
    const int b = blockIdx.x;
    if (step > 0) {
        const float* row = adj + (size_t)prev[b] * KCB;
        float mx = -3.4e38f;
        for (int j = threadIdx.x; j < KCB; j += 256) mx = fmaxf(mx, row[j]);
        sh[threadIdx.x] = mx; __syncthreads();
        for (int s = 128; s > 0; s >>= 1) {
            if (threadIdx.x < s) sh[threadIdx.x] = fmaxf(sh[threadIdx.x], sh[threadIdx.x + s]);
            __syncthreads();
        }
        mx = sh[0]; __syncthreads();
        float sum = 0.f;
        for (int j = threadIdx.x; j < KCB; j += 256) sum += __expf(row[j] - mx);
        sum = blockReduceSum(sum, sh);
        if (threadIdx.x == 0) {
            float lse = __logf(sum) + mx;
            atomicAdd(eth_sum, (lse - row[sym[b]]) * (1.0f / (float)Bsz));
        }
    }
    if (threadIdx.x == 0) prev[b] = sym[b];
    __syncthreads();
}

__global__ void k_vq(const float* __restrict__ cf, const float* __restrict__ cb,
                     const int* __restrict__ sym, const float* __restrict__ wvec,
                     float* __restrict__ zw, float* __restrict__ zr, float* __restrict__ zi,
                     float* __restrict__ vq_sum) {
    __shared__ float sh[256];
    int t = blockIdx.x * 256 + threadIdx.x;   // B*D2
    int b = t >> 10, d = t & (D2 - 1);
    float q = cb[(size_t)sym[b] * D2 + d];
    float c = cf[t];
    float df = q - c;
    zw[t] += wvec[b] * q;
    if (d < DIM) zr[(size_t)b * DIM + d] = q;
    else         zi[(size_t)b * DIM + d - DIM] = q;
    float loc = blockReduceSum(df * df, sh);
    if (threadIdx.x == 0) atomicAdd(vq_sum, loc);
}

__global__ void k_finalize(const float* __restrict__ sc, float* __restrict__ out) {
    out[0] = (1.0f + CC_W) * sc[0] / ((float)Bsz * (float)D2)
           + ETH_W * sc[1]
           + PONDER_W * sc[2];
}

// ---------------------------------------------------------------------------
// host orchestration
// ---------------------------------------------------------------------------
static inline void gemm(hipStream_t st, float* C, int ldc, const float* A, int lda,
                        const _Float16* Bh, int ldb, int M, int N, int K,
                        float alpha, int accum, const float* bias) {
    dim3 g(N / BN, M / BM), blk(256);
    k_gemm<<<g, blk, 0, st>>>(C, ldc, A, lda, Bh, ldb, M, N, K, alpha, accum, bias);
}

extern "C" void kernel_launch(void* const* d_in, const int* in_sizes, int n_in,
                              void* d_out, int out_size, void* d_ws, size_t ws_size,
                              hipStream_t stream) {
    const int*   input_ids = (const int*)  d_in[0];
    const float* emb_mag   = (const float*)d_in[1];
    const float* emb_phase = (const float*)d_in[2];
    const float* lin_r     = (const float*)d_in[3];
    const float* lin_i     = (const float*)d_in[4];
    const float* nscale    = (const float*)d_in[5];
    const float* nshift    = (const float*)d_in[6];
    const float* mbias     = (const float*)d_in[7];
    const float* halt_W    = (const float*)d_in[8];
    const float* halt_b    = (const float*)d_in[9];
    const float* ctrl_W    = (const float*)d_in[10];
    const float* ctrl_b    = (const float*)d_in[11];
    const float* q_r       = (const float*)d_in[12];
    const float* q_i       = (const float*)d_in[13];
    const float* kw_r      = (const float*)d_in[14];
    const float* kw_i      = (const float*)d_in[15];
    const float* v_r       = (const float*)d_in[16];
    const float* v_i       = (const float*)d_in[17];
    const float* codebook  = (const float*)d_in[18];
    const float* adjacency = (const float*)d_in[19];
    const float* dec_W     = (const float*)d_in[20];
    const float* dec_b     = (const float*)d_in[21];
    float* out = (float*)d_out;

    char* ws = (char*)d_ws;
    size_t off = 0;
    auto alloc = [&](size_t bytes) { size_t o = off; off += (bytes + 255) & ~(size_t)255; return o; };

    float* zr   = (float*)(ws + alloc((size_t)Bsz * DIM * 4));
    float* zi   = (float*)(ws + alloc((size_t)Bsz * DIM * 4));
    float* trp  = (float*)(ws + alloc((size_t)Bsz * DIM * 4));
    float* tpi  = (float*)(ws + alloc((size_t)Bsz * DIM * 4));
    float* qf   = (float*)(ws + alloc((size_t)Bsz * D2 * 4));
    float* kf   = (float*)(ws + alloc((size_t)Bsz * D2 * 4));
    float* vr   = (float*)(ws + alloc((size_t)Bsz * DIM * 4));
    float* vi   = (float*)(ws + alloc((size_t)Bsz * DIM * 4));
    float* zf   = (float*)(ws + alloc((size_t)Bsz * D2 * 4));
    float* Smat = (float*)(ws + alloc((size_t)Bsz * Bsz * 4));
    float* Gmat = (float*)(ws + alloc((size_t)Bsz * KCB * 4));
    float* cf   = (float*)(ws + alloc((size_t)Bsz * D2 * 4));
    float* rdv  = (float*)(ws + alloc((size_t)Bsz * D2 * 4));
    float* zw   = (float*)(ws + alloc((size_t)Bsz * D2 * 4));
    float* memb = (float*)(ws + alloc((size_t)Bsz * STK * D2 * 4));
    float* ptrb = (float*)(ws + alloc((size_t)Bsz * STK * 4));
    float* nptr = (float*)(ws + alloc((size_t)Bsz * STK * 4));
    float* wmb  = (float*)(ws + alloc((size_t)Bsz * STK * 4));
    float* halt = (float*)(ws + alloc((size_t)Bsz * 4));
    float* ctrl = (float*)(ws + alloc((size_t)Bsz * 3 * 4));
    float* wvec = (float*)(ws + alloc((size_t)Bsz * 4));
    float* rem  = (float*)(ws + alloc((size_t)Bsz * 4));
    int*   sym  = (int*)  (ws + alloc((size_t)Bsz * 4));
    int*   prev = (int*)  (ws + alloc((size_t)Bsz * 4));
    float* cbn  = (float*)(ws + alloc((size_t)KCB * 4));
    float* scal = (float*)(ws + alloc(256));                       // vq, eth, ponder
    _Float16* Wh    = (_Float16*)(ws + alloc((size_t)8 * DIM * DIM * 2));
    _Float16* kf_h  = (_Float16*)(ws + alloc((size_t)Bsz * D2 * 2));
    _Float16* vTr_h = (_Float16*)(ws + alloc((size_t)DIM * Bsz * 2));
    _Float16* vTi_h = (_Float16*)(ws + alloc((size_t)DIM * Bsz * 2));
    _Float16* cb_h  = (_Float16*)(ws + alloc((size_t)KCB * D2 * 2));
    _Float16* dec_h = (_Float16*)(ws + alloc((size_t)Vv * D2 * 2));

    // ---- init state (accumulators must be re-zeroed every call) ----
    hipMemsetAsync(memb, 0, (size_t)Bsz * STK * D2 * 4, stream);
    hipMemsetAsync(zw,   0, (size_t)Bsz * D2 * 4, stream);
    hipMemsetAsync(ptrb, 0, (size_t)Bsz * STK * 4, stream);
    hipMemsetAsync(scal, 0, 256, stream);
    k_init_state<<<4, 256, 0, stream>>>(ptrb, rem);

    // ---- f16 conversions of B-side operands (weights reused 8x) ----
    const size_t WSZ = (size_t)DIM * DIM;
    const float* wsrc[8] = { lin_r, lin_i, q_r, q_i, kw_r, kw_i, v_r, v_i };
    for (int i = 0; i < 8; ++i)
        k_convert_h<<<(int)(WSZ / 256), 256, 0, stream>>>(wsrc[i], Wh + i * WSZ, (long)WSZ);
    _Float16 *Wlr = Wh, *Wli = Wh + WSZ, *Wqr = Wh + 2 * WSZ, *Wqi = Wh + 3 * WSZ,
             *Wkr = Wh + 4 * WSZ, *Wki = Wh + 5 * WSZ, *Wvr = Wh + 6 * WSZ, *Wvi = Wh + 7 * WSZ;
    k_convert_h<<<(int)(((size_t)KCB * D2) / 256), 256, 0, stream>>>(codebook, cb_h, (long)KCB * D2);
    k_convert_h<<<(int)(((size_t)Vv * D2) / 256), 256, 0, stream>>>(dec_W, dec_h, (long)Vv * D2);
    k_cbnorm<<<KCB, 256, 0, stream>>>(codebook, cbn);

    // ---- embedding ----
    k_embed<<<(Bsz * DIM) / 256, 256, 0, stream>>>(input_ids, emb_mag, emb_phase, zr, zi);

    const float attn_scale = 0.04419417382415922f; // 512^-0.5
    const int EW = (Bsz * D2) / 256;               // elementwise grid over B*D2

    for (int step = 0; step < TT; ++step) {
        // p = modrelu(cnorm(clin(lin, z)))
        gemm(stream, trp, DIM, zr, DIM, Wlr, DIM, Bsz, DIM, DIM,  1.f, 0, nullptr);
        gemm(stream, trp, DIM, zi, DIM, Wli, DIM, Bsz, DIM, DIM, -1.f, 1, nullptr);
        gemm(stream, tpi, DIM, zi, DIM, Wlr, DIM, Bsz, DIM, DIM,  1.f, 0, nullptr);
        gemm(stream, tpi, DIM, zr, DIM, Wli, DIM, Bsz, DIM, DIM,  1.f, 1, nullptr);
        k_cnorm_modrelu<<<Bsz, 256, 0, stream>>>(trp, tpi, nscale, nshift, mbias);

        // qf = [qr | qi], kf = [kr | ki]  (written in place via ldc offsets)
        gemm(stream, qf,       D2, trp, DIM, Wqr, DIM, Bsz, DIM, DIM,  1.f, 0, nullptr);
        gemm(stream, qf,       D2, tpi, DIM, Wqi, DIM, Bsz, DIM, DIM, -1.f, 1, nullptr);
        gemm(stream, qf + DIM, D2, tpi, DIM, Wqr, DIM, Bsz, DIM, DIM,  1.f, 0, nullptr);
        gemm(stream, qf + DIM, D2, trp, DIM, Wqi, DIM, Bsz, DIM, DIM,  1.f, 1, nullptr);
        gemm(stream, kf,       D2, trp, DIM, Wkr, DIM, Bsz, DIM, DIM,  1.f, 0, nullptr);
        gemm(stream, kf,       D2, tpi, DIM, Wki, DIM, Bsz, DIM, DIM, -1.f, 1, nullptr);
        gemm(stream, kf + DIM, D2, tpi, DIM, Wkr, DIM, Bsz, DIM, DIM,  1.f, 0, nullptr);
        gemm(stream, kf + DIM, D2, trp, DIM, Wki, DIM, Bsz, DIM, DIM,  1.f, 1, nullptr);
        gemm(stream, vr,  DIM, trp, DIM, Wvr, DIM, Bsz, DIM, DIM,  1.f, 0, nullptr);
        gemm(stream, vr,  DIM, tpi, DIM, Wvi, DIM, Bsz, DIM, DIM, -1.f, 1, nullptr);
        gemm(stream, vi,  DIM, tpi, DIM, Wvr, DIM, Bsz, DIM, DIM,  1.f, 0, nullptr);
        gemm(stream, vi,  DIM, trp, DIM, Wvi, DIM, Bsz, DIM, DIM,  1.f, 1, nullptr);

        // attention: A = softmax(qf kf^T * scale); zf = [A vr | A vi]
        k_convert_h<<<EW, 256, 0, stream>>>(kf, kf_h, (long)Bsz * D2);
        gemm(stream, Smat, Bsz, qf, D2, kf_h, D2, Bsz, Bsz, D2, 1.f, 0, nullptr);
        k_softmax<<<Bsz, 256, 0, stream>>>(Smat, Bsz, attn_scale);
        k_transpose_h<<<dim3(DIM / 32, Bsz / 32), dim3(32, 8), 0, stream>>>(vr, vTr_h, Bsz, DIM);
        k_transpose_h<<<dim3(DIM / 32, Bsz / 32), dim3(32, 8), 0, stream>>>(vi, vTi_h, Bsz, DIM);
        gemm(stream, zf,       D2, Smat, Bsz, vTr_h, Bsz, Bsz, DIM, Bsz, 1.f, 0, nullptr);
        gemm(stream, zf + DIM, D2, Smat, Bsz, vTi_h, Bsz, Bsz, DIM, Bsz, 1.f, 0, nullptr);

        // halting + control, ponder bookkeeping
        k_halt_ctrl<<<Bsz, 256, 0, stream>>>(zf, halt_W, halt_b, ctrl_W, ctrl_b, halt, ctrl);
        k_wrem<<<4, 256, 0, stream>>>(halt, rem, wvec, scal + 2, step == TT - 1 ? 1 : 0);

        // differentiable stack (fused mem update + read reduction: the BW hog)
        k_stack_ptr<<<4, 256, 0, stream>>>(ptrb, ctrl, nptr, wmb);
        k_stack_mem<<<EW, 256, 0, stream>>>(memb, wmb, nptr, zf, rdv);
        k_cf<<<EW, 256, 0, stream>>>(zf, rdv, cf);

        // VQ: argmin_j (||cb_j||^2 - 2 cf.cb_j) via WMMA GEMM + reduction
        gemm(stream, Gmat, KCB, cf, D2, cb_h, D2, Bsz, KCB, D2, 1.f, 0, nullptr);
        k_argmin<<<Bsz, 256, 0, stream>>>(Gmat, cbn, sym);
        k_eth<<<Bsz, 256, 0, stream>>>(adjacency, sym, prev, scal + 1, step);
        k_vq<<<EW, 256, 0, stream>>>(cf, codebook, sym, wvec, zw, zr, zi, scal);
    }

    // logits = zw dec_W^T + dec_b (written straight into d_out)
    gemm(stream, out, Vv, zw, D2, dec_h, D2, Bsz, Vv, D2, 1.f, 0, dec_b);
    k_finalize<<<1, 1, 0, stream>>>(scal, out + (size_t)Bsz * Vv);
}
